// AttentionBridge_72825465471231
// MI455X (gfx1250) — matrix operand order repaired and checked by reading, not compiled, not run on hardware
//
#include <hip/hip_runtime.h>
#include <math.h>

// Problem constants (from reference)
#define B_   2
#define LD_  512
#define LE_  2048
#define D_   1024
#define H_   16
#define HD_  64
#define ZB_  (B_*H_)     // 32 batched (b,h) slices
#define SCALE_ 0.125f    // 64^-0.5
#define NS_  38          // N_SAMPLE = int(5*ln(2049))
#define NT_  31          // N_TOP    = int(5*ln(513))
#define WLOC_ 512        // local window = max(4, L_ENC/4)

typedef __attribute__((ext_vector_type(16))) __bf16 v16bf;
typedef __attribute__((ext_vector_type(8)))  float  v8f;

union ABu { v16bf v; unsigned int u[8]; };

// gfx1250 async load-to-LDS path (ASYNCcnt + s_wait_asynccnt), with fallback.
#if __has_builtin(__builtin_amdgcn_global_load_async_to_lds_b128) && \
    __has_builtin(__builtin_amdgcn_s_wait_asynccnt)
#define ASYNC_LDS 1
#else
#define ASYNC_LDS 0
#endif

#if ASYNC_LDS
// Builtin signature (from clang diagnostic): param0 is AS1 pointer to a
// 16-byte int vector, param1 the LDS-side (AS3) pointer, then imm offset+cpol.
typedef __attribute__((vector_size(4 * sizeof(int)))) int afi4;
__device__ __forceinline__ void async_ld16(const void* g, void* l) {
  __builtin_amdgcn_global_load_async_to_lds_b128(
      (__attribute__((address_space(1))) afi4*)g,
      (__attribute__((address_space(3))) afi4*)l, 0, 0);
}
#endif

__device__ __forceinline__ unsigned short f2bf(float f) {
  unsigned int u = __float_as_uint(f);
  u += 0x7FFFu + ((u >> 16) & 1u);          // round-to-nearest-even
  return (unsigned short)(u >> 16);
}
__device__ __forceinline__ float bf2f(unsigned short h) {
  return __uint_as_float(((unsigned int)h) << 16);
}
__device__ __forceinline__ float eluf(float x) { return x > 0.f ? x : (expf(x) - 1.f); }
__device__ __forceinline__ int imin(int a, int b) { return a < b ? a : b; }
__device__ __forceinline__ int imax(int a, int b) { return a > b ? a : b; }

__device__ float blockSum(float v, float* red) {
  int tid = threadIdx.x;
  red[tid] = v; __syncthreads();
  for (int off = blockDim.x >> 1; off > 0; off >>= 1) {
    if (tid < off) red[tid] += red[tid + off];
    __syncthreads();
  }
  float r = red[0]; __syncthreads(); return r;
}
__device__ float blockMax(float v, float* red) {
  int tid = threadIdx.x;
  red[tid] = v; __syncthreads();
  for (int off = blockDim.x >> 1; off > 0; off >>= 1) {
    if (tid < off) red[tid] = fmaxf(red[tid], red[tid + off]);
    __syncthreads();
  }
  float r = red[0]; __syncthreads(); return r;
}

// ---------------------------------------------------------------------------
// Batched WMMA GEMM:  C[z][m,n] = sum_k A[z][m,k] * Bt[z][n,k]
// Both operands bf16, row-major over K.  Workgroup = 128 threads (4 waves),
// computes a 64x64 C tile; each wave owns a 32x32 sub-tile as 2x2 WMMA
// accumulators, so each A/B fragment pair feeds 4 v_wmma_f32_16x16x32_bf16.
// K is consumed in 32-deep panels staged into double-buffered LDS
// (async global->LDS on gfx1250, ASYNCcnt-tracked; plain ld/st fallback).
// Requires M%64==0, N%64==0, K%32==0 (true for every call here).
// Fragment gathers follow ISA 7.12.2:
//   A 16x32: lane m=l&15, half=l>>4; dwords v0..3 at K bytes half*16,
//            v4..7 at 32+half*16 (within a 64-byte K row)
//   B 32x16: lane n=l&15, half=l>>4; dwords v0..3 at K bytes half*16,
//            v4..7 at 32+half*16 (same K order as A)
//   C 16x16: lane n=l&15; VGPR r holds row m=(l>>4)*8+r
// ---------------------------------------------------------------------------
__global__ void __launch_bounds__(128)
k_wmma_gemm(const unsigned short* __restrict__ A,
            const unsigned short* __restrict__ Bt,
            void* __restrict__ Cv,
            int M, int N, int K, int lda, int ldb,
            long long ldcM, long long ldcN, int Hs,
            long long sAb, long long sAh,
            long long sBb, long long sBh,
            long long sCb, long long sCh, int outBf) {
  __shared__ __align__(16) unsigned short sA[2][64 * 32];
  __shared__ __align__(16) unsigned short sB[2][64 * 32];

  int z  = blockIdx.z;
  int zb = z / Hs, zh = z % Hs;
  const unsigned short* Ab = A  + zb * sAb + zh * sAh;
  const unsigned short* Bb = Bt + zb * sBb + zh * sBh;
  long long cOff = zb * sCb + zh * sCh;

  int tid  = threadIdx.x;       // 0..127
  int wave = tid >> 5;
  int lane = tid & 31;
  int lm   = lane & 15;
  int half = lane >> 4;
  int wm   = wave >> 1;         // wave's 32-row block within 64
  int wn   = wave & 1;          // wave's 32-col block within 64
  int m0 = blockIdx.y * 64;
  int n0 = blockIdx.x * 64;

  // Staging map: panel = 64 rows x 64 bytes; 256 16B chunks, 2 per thread.
  int ch0 = tid * 2, ch1 = tid * 2 + 1;
  int rA0 = ch0 >> 2, sg0 = (ch0 & 3) * 16;
  int rA1 = ch1 >> 2, sg1 = (ch1 & 3) * 16;

  auto stage = [&](int kt, int buf) {
    long long kB = (long long)kt * 32 * 2;  // byte offset along K
    const char* gA0 = (const char*)Ab + ((long long)(m0 + rA0) * lda) * 2 + kB + sg0;
    const char* gA1 = (const char*)Ab + ((long long)(m0 + rA1) * lda) * 2 + kB + sg1;
    const char* gB0 = (const char*)Bb + ((long long)(n0 + rA0) * ldb) * 2 + kB + sg0;
    const char* gB1 = (const char*)Bb + ((long long)(n0 + rA1) * ldb) * 2 + kB + sg1;
    char* lA0 = (char*)sA[buf] + rA0 * 64 + sg0;
    char* lA1 = (char*)sA[buf] + rA1 * 64 + sg1;
    char* lB0 = (char*)sB[buf] + rA0 * 64 + sg0;
    char* lB1 = (char*)sB[buf] + rA1 * 64 + sg1;
#if ASYNC_LDS
    async_ld16(gA0, lA0);
    async_ld16(gA1, lA1);
    async_ld16(gB0, lB0);
    async_ld16(gB1, lB1);
#else
    *(int4*)lA0 = *(const int4*)gA0;
    *(int4*)lA1 = *(const int4*)gA1;
    *(int4*)lB0 = *(const int4*)gB0;
    *(int4*)lB1 = *(const int4*)gB1;
#endif
  };

  v8f acc00 = {}, acc01 = {}, acc10 = {}, acc11 = {};
  int nk = K / 32;

  stage(0, 0);
  for (int kt = 0; kt < nk; ++kt) {
    int buf = kt & 1;
    if (kt + 1 < nk) stage(kt + 1, buf ^ 1);
#if ASYNC_LDS
    if (kt + 1 < nk) __builtin_amdgcn_s_wait_asynccnt(4);
    else             __builtin_amdgcn_s_wait_asynccnt(0);
#endif
    __syncthreads();

    ABu a0, a1, b0, b1;
    {
      const char* pa0 = (const char*)sA[buf] + (wm * 32 + lm) * 64 + half * 16;
      *(int4*)&a0.u[0] = *(const int4*)pa0;
      *(int4*)&a0.u[4] = *(const int4*)(pa0 + 32);
      const char* pa1 = (const char*)sA[buf] + (wm * 32 + 16 + lm) * 64 + half * 16;
      *(int4*)&a1.u[0] = *(const int4*)pa1;
      *(int4*)&a1.u[4] = *(const int4*)(pa1 + 32);
      const char* pb0 = (const char*)sB[buf] + (wn * 32 + lm) * 64 + half * 16;
      *(int4*)&b0.u[0] = *(const int4*)pb0;
      *(int4*)&b0.u[4] = *(const int4*)(pb0 + 32);
      const char* pb1 = (const char*)sB[buf] + (wn * 32 + 16 + lm) * 64 + half * 16;
      *(int4*)&b1.u[0] = *(const int4*)pb1;
      *(int4*)&b1.u[4] = *(const int4*)(pb1 + 32);
    }
    acc00 = __builtin_amdgcn_wmma_f32_16x16x32_bf16(false, a0.v, false, b0.v, (short)0, acc00, false, false);
    acc01 = __builtin_amdgcn_wmma_f32_16x16x32_bf16(false, a0.v, false, b1.v, (short)0, acc01, false, false);
    acc10 = __builtin_amdgcn_wmma_f32_16x16x32_bf16(false, a1.v, false, b0.v, (short)0, acc10, false, false);
    acc11 = __builtin_amdgcn_wmma_f32_16x16x32_bf16(false, a1.v, false, b1.v, (short)0, acc11, false, false);
    __syncthreads();
  }

  // Epilogue: 4 sub-tiles of 16x16
  v8f* accs[4] = { &acc00, &acc01, &acc10, &acc11 };
#pragma unroll
  for (int sm = 0; sm < 2; ++sm) {
#pragma unroll
    for (int sn = 0; sn < 2; ++sn) {
      v8f& acc = *accs[sm * 2 + sn];
      long long mb = m0 + wm * 32 + sm * 16 + half * 8;
      long long nb = n0 + wn * 32 + sn * 16 + lm;
#pragma unroll
      for (int r = 0; r < 8; ++r) {
        long long idx = cOff + (mb + r) * ldcM + nb * ldcN;
        if (outBf) ((unsigned short*)Cv)[idx] = f2bf(acc[r]);
        else       ((float*)Cv)[idx] = acc[r];
      }
    }
  }
}

// --------------------------- elementwise / utility --------------------------

__global__ void k_f2bf(const float* __restrict__ x, unsigned short* __restrict__ y, int n) {
  int id = blockIdx.x * 256 + threadIdx.x;
  if (id < n) y[id] = f2bf(x[id]);
}

// Wt[n,k] = W[k,n]  (so X@W becomes A·Wtᵀ)
__global__ void k_transpose_w(const float* __restrict__ W, unsigned short* __restrict__ Wt) {
  int id = blockIdx.x * 256 + threadIdx.x;
  if (id >= D_ * D_) return;
  int k = id / D_, n = id % D_;
  Wt[(long long)n * D_ + k] = f2bf(W[id]);
}

// Vt[(b,h,d),s] = V[(b,s),(h,d)]
__global__ void k_transpose_v(const unsigned short* __restrict__ V, unsigned short* __restrict__ Vt) {
  int id = blockIdx.x * 256 + threadIdx.x;
  if (id >= B_ * LE_ * D_) return;
  int c = id % D_;
  int r = id / D_;
  int b = r / LE_, s = r % LE_;
  int h = c / HD_, d = c % HD_;
  Vt[((long long)(b * H_ + h) * HD_ + d) * LE_ + s] = V[id];
}

__global__ void k_alpha(const float* __restrict__ a, float* __restrict__ w) {
  if (blockIdx.x == 0 && threadIdx.x == 0) {
    float mx = a[0];
    for (int i = 1; i < 6; ++i) mx = fmaxf(mx, a[i]);
    float e[6], s = 0.f;
    for (int i = 0; i < 6; ++i) { e[i] = expf(a[i] - mx); s += e[i]; }
    for (int i = 0; i < 6; ++i) w[i] = e[i] / s;
  }
}

// ------------------------------- softmaxes ----------------------------------

__global__ void k_softmax_sdp(const float* __restrict__ S, unsigned short* __restrict__ P) {
  int t = blockIdx.x, z = blockIdx.y;
  long long base = ((long long)z * LD_ + t) * LE_;
  const float* row = S + base;
  unsigned short* pr = P + base;
  __shared__ float red[256];
  float mx = -3.4e38f;
  for (int s = threadIdx.x; s < LE_; s += 256) mx = fmaxf(mx, row[s] * SCALE_);
  mx = blockMax(mx, red);
  float sm = 0.f;
  for (int s = threadIdx.x; s < LE_; s += 256) sm += expf(row[s] * SCALE_ - mx);
  sm = blockSum(sm, red);
  float inv = 1.f / sm;
  for (int s = threadIdx.x; s < LE_; s += 256)
    pr[s] = f2bf(expf(row[s] * SCALE_ - mx) * inv);
}

__global__ void k_softmax_loc(const float* __restrict__ S, unsigned short* __restrict__ P) {
  int t = blockIdx.x, z = blockIdx.y;
  long long base = ((long long)z * LD_ + t) * LE_;
  const float* row = S + base;
  unsigned short* pr = P + base;
  int center = (int)((float)t * ((float)LE_ / (float)LD_));
  center = imin(imax(center, 0), LE_ - 1);
  int lo = imax(center - WLOC_ / 2, 0);
  int hi = imin(center + WLOC_ / 2, LE_ - 1);
  __shared__ float red[256];
  float mx = -3.4e38f;
  for (int s = threadIdx.x; s < LE_; s += 256)
    if (s >= lo && s <= hi) mx = fmaxf(mx, row[s] * SCALE_);
  mx = blockMax(mx, red);
  float sm = 0.f;
  for (int s = threadIdx.x; s < LE_; s += 256)
    if (s >= lo && s <= hi) sm += expf(row[s] * SCALE_ - mx);
  sm = blockSum(sm, red);
  float inv = 1.f / sm;
  for (int s = threadIdx.x; s < LE_; s += 256)
    pr[s] = (s >= lo && s <= hi) ? f2bf(expf(row[s] * SCALE_ - mx) * inv)
                                 : (unsigned short)0;
}

// ------------------------- linear / cosine features -------------------------

__global__ void k_featlin_q(const unsigned short* __restrict__ Qbf, unsigned short* __restrict__ qf) {
  int id = blockIdx.x * 256 + threadIdx.x;
  if (id >= ZB_ * LD_ * HD_) return;
  int d = id % HD_, t = (id / HD_) % LD_, z = id / (HD_ * LD_);
  int zb = z / H_, zh = z % H_;
  float q = bf2f(Qbf[((long long)(zb * LD_ + t)) * D_ + zh * HD_ + d]);
  qf[id] = f2bf(eluf(q * SCALE_) + 1.f);
}

__global__ void k_featlin_k(const unsigned short* __restrict__ Kbf, unsigned short* __restrict__ kfT) {
  int id = blockIdx.x * 256 + threadIdx.x;
  if (id >= ZB_ * HD_ * LE_) return;
  int s = id % LE_, d = (id / LE_) % HD_, z = id / (LE_ * HD_);
  int zb = z / H_, zh = z % H_;
  float k = bf2f(Kbf[((long long)(zb * LE_ + s)) * D_ + zh * HD_ + d]);
  kfT[id] = f2bf(eluf(k) + 1.f);
}

__global__ void k_cos_q(const unsigned short* __restrict__ Qbf, unsigned short* __restrict__ qc) {
  int id = blockIdx.x * 256 + threadIdx.x;
  if (id >= ZB_ * LD_) return;
  int z = id / LD_, t = id % LD_;
  int zb = z / H_, zh = z % H_;
  const unsigned short* q = Qbf + ((long long)(zb * LD_ + t)) * D_ + zh * HD_;
  float ss = 0.f;
  for (int d = 0; d < HD_; ++d) { float x = bf2f(q[d]); ss += x * x; }
  float inv = 1.f / fmaxf(sqrtf(ss), 1e-12f);
  for (int d = 0; d < HD_; ++d) {
    float x = bf2f(q[d]) * inv;
    qc[(long long)id * HD_ + d] = f2bf(fmaxf(x, 0.f) + 1e-6f);
  }
}

__global__ void k_cos_k(const unsigned short* __restrict__ Kbf, unsigned short* __restrict__ kcT) {
  int id = blockIdx.x * 256 + threadIdx.x;
  if (id >= ZB_ * LE_) return;
  int z = id / LE_, s = id % LE_;
  int zb = z / H_, zh = z % H_;
  const unsigned short* k = Kbf + ((long long)(zb * LE_ + s)) * D_ + zh * HD_;
  float ss = 0.f;
  for (int d = 0; d < HD_; ++d) { float x = bf2f(k[d]); ss += x * x; }
  float inv = 1.f / fmaxf(sqrtf(ss), 1e-12f);
  for (int d = 0; d < HD_; ++d) {
    float x = bf2f(k[d]) * inv;
    kcT[((long long)z * HD_ + d) * LE_ + s] = f2bf(fmaxf(x, 0.f) + 1e-6f);
  }
}

__global__ void k_rowsum_bf(const unsigned short* __restrict__ X, float* __restrict__ out,
                            int len, float scale) {
  int row = blockIdx.x;
  __shared__ float red[256];
  const unsigned short* xr = X + (long long)row * len;
  float s = 0.f;
  for (int i = threadIdx.x; i < len; i += 256) s += bf2f(xr[i]);
  s = blockSum(s, red);
  if (threadIdx.x == 0) out[row] = s * scale;
}

__global__ void k_div_denom(const unsigned short* __restrict__ qf, const float* __restrict__ ksum,
                            float* __restrict__ O) {
  int id = blockIdx.x * 256 + threadIdx.x;
  if (id >= ZB_ * LD_) return;
  int z = id / LD_;
  const unsigned short* q = qf + (long long)id * HD_;
  const float* ks = ksum + z * HD_;
  float den = 0.f;
  for (int d = 0; d < HD_; ++d) den += bf2f(q[d]) * ks[d];
  float inv = 1.f / fmaxf(den, 1e-6f);
  float* o = O + (long long)id * HD_;
  for (int v = 0; v < HD_; ++v) o[v] *= inv;
}

// -------------------------------- probsparse --------------------------------

__global__ void k_ps_meas(const unsigned short* __restrict__ Qbf, const unsigned short* __restrict__ Kbf,
                          const int* __restrict__ sidx, float* __restrict__ Mval) {
  int id = blockIdx.x * 256 + threadIdx.x;
  if (id >= ZB_ * LD_) return;
  int z = id / LD_, t = id % LD_;
  int zb = z / H_, zh = z % H_;
  const unsigned short* q = Qbf + ((long long)(zb * LD_ + t)) * D_ + zh * HD_;
  float mx = -3.4e38f, sm = 0.f;
  for (int j = 0; j < NS_; ++j) {
    int s = sidx[j];
    const unsigned short* kr = Kbf + ((long long)(zb * LE_ + s)) * D_ + zh * HD_;
    float dot = 0.f;
    for (int d = 0; d < HD_; ++d) dot += bf2f(q[d]) * bf2f(kr[d]);
    dot *= SCALE_;
    mx = fmaxf(mx, dot); sm += dot;
  }
  Mval[id] = mx - sm / (float)NS_;
}

__global__ void k_topk(const float* __restrict__ Mval, int* __restrict__ topidx) {
  int z = blockIdx.x, tid = threadIdx.x;
  __shared__ float vals[LD_];
  __shared__ float rv[256];
  __shared__ int   ri[256];
  for (int i = tid; i < LD_; i += 256) vals[i] = Mval[z * LD_ + i];
  __syncthreads();
  for (int it = 0; it < NT_; ++it) {
    float best = -3.4e38f; int bi = 0;
    for (int i = tid; i < LD_; i += 256)
      if (vals[i] > best) { best = vals[i]; bi = i; }
    rv[tid] = best; ri[tid] = bi; __syncthreads();
    for (int off = 128; off > 0; off >>= 1) {
      if (tid < off && rv[tid + off] > rv[tid]) { rv[tid] = rv[tid + off]; ri[tid] = ri[tid + off]; }
      __syncthreads();
    }
    if (tid == 0) { topidx[z * NT_ + it] = ri[0]; vals[ri[0]] = -3.4e38f; }
    __syncthreads();
  }
}

__global__ void k_fill_vmean(const float* __restrict__ vmean, float* __restrict__ O) {
  int id = blockIdx.x * 256 + threadIdx.x;
  if (id >= ZB_ * LD_ * HD_) return;
  int d = id % HD_, z = id / (LD_ * HD_);
  O[id] = vmean[z * HD_ + d];
}

__global__ void k_toprow(const unsigned short* __restrict__ Qbf, const unsigned short* __restrict__ Kbf,
                         const unsigned short* __restrict__ Vtbf, const int* __restrict__ topidx,
                         float* __restrict__ O) {
  int i = blockIdx.x, z = blockIdx.y, tid = threadIdx.x;
  int zb = z / H_, zh = z % H_;
  int t = topidx[z * NT_ + i];
  __shared__ float sc[LE_];
  __shared__ float red[256];
  __shared__ float qs[HD_];
  if (tid < HD_) qs[tid] = bf2f(Qbf[((long long)(zb * LD_ + t)) * D_ + zh * HD_ + tid]);
  __syncthreads();
  float mx = -3.4e38f;
  for (int s = tid; s < LE_; s += 256) {
    const unsigned short* kr = Kbf + ((long long)(zb * LE_ + s)) * D_ + zh * HD_;
    float dot = 0.f;
    for (int d = 0; d < HD_; ++d) dot += qs[d] * bf2f(kr[d]);
    dot *= SCALE_;
    sc[s] = dot;
    mx = fmaxf(mx, dot);
  }
  mx = blockMax(mx, red);
  float sm = 0.f;
  for (int s = tid; s < LE_; s += 256) { float e = expf(sc[s] - mx); sc[s] = e; sm += e; }
  sm = blockSum(sm, red);
  float inv = 1.f / sm;
  for (int d = 0; d < HD_; ++d) {
    const unsigned short* vr = Vtbf + ((long long)z * HD_ + d) * LE_;
    float part = 0.f;
    for (int s = tid; s < LE_; s += 256) part += sc[s] * bf2f(vr[s]);
    part = blockSum(part, red);
    if (tid == 0) O[((long long)z * LD_ + t) * HD_ + d] = part * inv;
  }
}

// --------------------------- merge + final mixing ---------------------------

__global__ void k_merge(const float* __restrict__ O, unsigned short* __restrict__ Y) {
  int id = blockIdx.x * 256 + threadIdx.x;
  if (id >= B_ * LD_ * D_) return;
  int c = id % D_, r = id / D_;
  int b = r / LD_, t = r % LD_;
  int h = c / HD_, d = c % HD_;
  Y[id] = f2bf(O[((long long)(b * H_ + h) * LD_ + t) * HD_ + d]);
}

__global__ void k_finalize(const float* __restrict__ dec,
                           const float* __restrict__ Z0, const float* __restrict__ Z1,
                           const float* __restrict__ Z2, const float* __restrict__ Z3,
                           const float* __restrict__ Z4,
                           const float* __restrict__ rmsw, const float* __restrict__ w,
                           float* __restrict__ out) {
  int r = blockIdx.x;
  const float* Zs[5] = { Z0, Z1, Z2, Z3, Z4 };
  __shared__ float red[256];
  __shared__ float msv[5];
  long long base = (long long)r * D_;
  for (int i = 0; i < 5; ++i) {
    float ss = 0.f;
    for (int c = threadIdx.x; c < D_; c += 256) {
      float v = dec[base + c] + Zs[i][base + c];
      ss += v * v;
    }
    ss = blockSum(ss, red);
    if (threadIdx.x == 0) msv[i] = rsqrtf(ss / (float)D_ + 1e-6f);
    __syncthreads();
  }
  float w0 = w[0];
  for (int c = threadIdx.x; c < D_; c += 256) {
    float d0 = dec[base + c];
    float acc = w0 * d0;
    for (int i = 0; i < 5; ++i)
      acc += w[i + 1] * (d0 + Zs[i][base + c]) * msv[i] * rmsw[c];
    out[base + c] = acc;
  }
}

// ---------------------------------------------------------------------------

extern "C" void kernel_launch(void* const* d_in, const int* in_sizes, int n_in,
                              void* d_out, int out_size, void* d_ws, size_t ws_size,
                              hipStream_t stream) {
  (void)in_sizes; (void)n_in; (void)out_size; (void)ws_size;
  const float* dec  = (const float*)d_in[0];
  const float* enc  = (const float*)d_in[1];
  const float* Wq   = (const float*)d_in[2];
  const float* Wk   = (const float*)d_in[3];
  const float* Wv   = (const float*)d_in[4];
  const float* Wo   = (const float*)d_in[5];
  const float* rmsw = (const float*)d_in[6];
  const float* alph = (const float*)d_in[7];
  const int*   sidx = (const int*)d_in[8];
  float* out = (float*)d_out;

  // Workspace carve-out (requires ~400 MB of scratch).
  char* p = (char*)d_ws;
  size_t off = 0;
  auto alloc = [&](size_t bytes) -> void* {
    void* r = p + off;
    off = (off + bytes + 255) & ~(size_t)255;
    return r;
  };
  typedef unsigned short us;
  float* w6      = (float*)alloc(6 * 4);
  us* Xdec_bf    = (us*)alloc((size_t)B_ * LD_ * D_ * 2);
  us* Xenc_bf    = (us*)alloc((size_t)B_ * LE_ * D_ * 2);
  us* WqT        = (us*)alloc((size_t)D_ * D_ * 2);
  us* WkT        = (us*)alloc((size_t)D_ * D_ * 2);
  us* WvT        = (us*)alloc((size_t)D_ * D_ * 2);
  us* WoT        = (us*)alloc((size_t)D_ * D_ * 2);
  us* Qbf        = (us*)alloc((size_t)B_ * LD_ * D_ * 2);
  us* Kbf        = (us*)alloc((size_t)B_ * LE_ * D_ * 2);
  us* Vbf        = (us*)alloc((size_t)B_ * LE_ * D_ * 2);
  us* Vtbf       = (us*)alloc((size_t)ZB_ * HD_ * LE_ * 2);
  float* S       = (float*)alloc((size_t)ZB_ * LD_ * LE_ * 4);   // 128 MB
  us* Psdp       = (us*)alloc((size_t)ZB_ * LD_ * LE_ * 2);      // 64 MB
  us* Ploc       = (us*)alloc((size_t)ZB_ * LD_ * LE_ * 2);      // 64 MB
  float* O0      = (float*)alloc((size_t)ZB_ * LD_ * HD_ * 4);
  float* O1      = (float*)alloc((size_t)ZB_ * LD_ * HD_ * 4);
  float* O2      = (float*)alloc((size_t)ZB_ * LD_ * HD_ * 4);
  float* O3      = (float*)alloc((size_t)ZB_ * LD_ * HD_ * 4);
  float* O4      = (float*)alloc((size_t)ZB_ * LD_ * HD_ * 4);
  us* qf         = (us*)alloc((size_t)ZB_ * LD_ * HD_ * 2);
  us* kfT        = (us*)alloc((size_t)ZB_ * HD_ * LE_ * 2);
  float* ksum    = (float*)alloc((size_t)ZB_ * HD_ * 4);
  us* kvT        = (us*)alloc((size_t)ZB_ * HD_ * HD_ * 2);
  us* qc         = (us*)alloc((size_t)ZB_ * LD_ * HD_ * 2);
  us* kcT        = (us*)alloc((size_t)ZB_ * HD_ * LE_ * 2);
  float* ksumc   = (float*)alloc((size_t)ZB_ * HD_ * 4);
  us* kvcT       = (us*)alloc((size_t)ZB_ * HD_ * HD_ * 2);
  float* Mval    = (float*)alloc((size_t)ZB_ * LD_ * 4);
  int* topidx    = (int*)alloc((size_t)ZB_ * NT_ * 4);
  float* vmean   = (float*)alloc((size_t)ZB_ * HD_ * 4);
  us* Ybf        = (us*)alloc((size_t)B_ * LD_ * D_ * 2);
  float* Zp[5];
  for (int i = 0; i < 5; ++i) Zp[i] = (float*)alloc((size_t)B_ * LD_ * D_ * 4);

  auto cdiv = [](long long n, int b) { return (int)((n + b - 1) / b); };
  auto gemm = [&](const us* A, const us* Bt, void* C, int M, int N, int K,
                  int lda, int ldb, long long ldcM, long long ldcN,
                  int Zn, int Hs, long long sAb, long long sAh,
                  long long sBb, long long sBh, long long sCb, long long sCh, int obf) {
    dim3 g(N / 64, M / 64, Zn);
    k_wmma_gemm<<<g, dim3(128), 0, stream>>>(A, Bt, C, M, N, K, lda, ldb,
                                             ldcM, ldcN, Hs, sAb, sAh, sBb, sBh, sCb, sCh, obf);
  };

  // ---- input conversion / weight transposition ----
  k_f2bf<<<cdiv((long long)B_ * LD_ * D_, 256), 256, 0, stream>>>(dec, Xdec_bf, B_ * LD_ * D_);
  k_f2bf<<<cdiv((long long)B_ * LE_ * D_, 256), 256, 0, stream>>>(enc, Xenc_bf, B_ * LE_ * D_);
  k_transpose_w<<<cdiv((long long)D_ * D_, 256), 256, 0, stream>>>(Wq, WqT);
  k_transpose_w<<<cdiv((long long)D_ * D_, 256), 256, 0, stream>>>(Wk, WkT);
  k_transpose_w<<<cdiv((long long)D_ * D_, 256), 256, 0, stream>>>(Wv, WvT);
  k_transpose_w<<<cdiv((long long)D_ * D_, 256), 256, 0, stream>>>(Wo, WoT);
  k_alpha<<<1, 32, 0, stream>>>(alph, w6);

  // ---- projections (WMMA) ----
  gemm(Xdec_bf, WqT, Qbf, B_ * LD_, D_, D_, D_, D_, D_, 1, 1, 1, 0, 0, 0, 0, 0, 0, 1);
  gemm(Xenc_bf, WkT, Kbf, B_ * LE_, D_, D_, D_, D_, D_, 1, 1, 1, 0, 0, 0, 0, 0, 0, 1);
  gemm(Xenc_bf, WvT, Vbf, B_ * LE_, D_, D_, D_, D_, D_, 1, 1, 1, 0, 0, 0, 0, 0, 0, 1);
  k_transpose_v<<<cdiv((long long)B_ * LE_ * D_, 256), 256, 0, stream>>>(Vbf, Vtbf);

  const long long zQ = (long long)LD_ * D_, zK = (long long)LE_ * D_;
  const long long zS = (long long)LD_ * LE_, zV = (long long)HD_ * LE_;
  const long long zO = (long long)LD_ * HD_, zKV = (long long)HD_ * HD_;

  // ---- shared score matrix S = q·kᵀ (un-scaled), batched over (b,h) ----
  gemm(Qbf, Kbf, S, LD_, LE_, HD_, D_, D_, LE_, 1,
       ZB_, H_, zQ, HD_, zK, HD_, (long long)H_ * zS, zS, 0);

  // ---- branch 0: SDP; branch 4: local (both reuse S) ----
  k_softmax_sdp<<<dim3(LD_, ZB_), 256, 0, stream>>>(S, Psdp);
  k_softmax_loc<<<dim3(LD_, ZB_), 256, 0, stream>>>(S, Ploc);
  gemm(Psdp, Vtbf, O0, LD_, HD_, LE_, LE_, LE_, HD_, 1,
       ZB_, H_, (long long)H_ * zS, zS, (long long)H_ * zV, zV, (long long)H_ * zO, zO, 0);
  gemm(Ploc, Vtbf, O4, LD_, HD_, LE_, LE_, LE_, HD_, 1,
       ZB_, H_, (long long)H_ * zS, zS, (long long)H_ * zV, zV, (long long)H_ * zO, zO, 0);

  // ---- branch 1: linear attention ----
  k_featlin_q<<<cdiv((long long)ZB_ * LD_ * HD_, 256), 256, 0, stream>>>(Qbf, qf);
  k_featlin_k<<<cdiv((long long)ZB_ * HD_ * LE_, 256), 256, 0, stream>>>(Kbf, kfT);
  k_rowsum_bf<<<ZB_ * HD_, 256, 0, stream>>>(kfT, ksum, LE_, 1.f);
  gemm(Vtbf, kfT, kvT, HD_, HD_, LE_, LE_, LE_, HD_, 1,
       ZB_, H_, (long long)H_ * zV, zV, (long long)H_ * zV, zV, (long long)H_ * zKV, zKV, 1);
  gemm(qf, kvT, O1, LD_, HD_, HD_, HD_, HD_, HD_, 1,
       ZB_, H_, (long long)H_ * zO, zO, (long long)H_ * zKV, zKV, (long long)H_ * zO, zO, 0);
  k_div_denom<<<cdiv((long long)ZB_ * LD_, 256), 256, 0, stream>>>(qf, ksum, O1);

  // ---- branch 3: cosine attention ----
  k_cos_q<<<cdiv((long long)ZB_ * LD_, 256), 256, 0, stream>>>(Qbf, qc);
  k_cos_k<<<cdiv((long long)ZB_ * LE_, 256), 256, 0, stream>>>(Kbf, kcT);
  k_rowsum_bf<<<ZB_ * HD_, 256, 0, stream>>>(kcT, ksumc, LE_, 1.f);
  gemm(Vtbf, kcT, kvcT, HD_, HD_, LE_, LE_, LE_, HD_, 1,
       ZB_, H_, (long long)H_ * zV, zV, (long long)H_ * zV, zV, (long long)H_ * zKV, zKV, 1);
  gemm(qc, kvcT, O3, LD_, HD_, HD_, HD_, HD_, HD_, 1,
       ZB_, H_, (long long)H_ * zO, zO, (long long)H_ * zKV, zKV, (long long)H_ * zO, zO, 0);
  k_div_denom<<<cdiv((long long)ZB_ * LD_, 256), 256, 0, stream>>>(qc, ksumc, O3);

  // ---- branch 2: probsparse ----
  k_ps_meas<<<cdiv((long long)ZB_ * LD_, 256), 256, 0, stream>>>(Qbf, Kbf, sidx, Mval);
  k_topk<<<ZB_, 256, 0, stream>>>(Mval, topidx);
  k_rowsum_bf<<<ZB_ * HD_, 256, 0, stream>>>(Vtbf, vmean, LE_, 1.f / (float)LE_);
  k_fill_vmean<<<cdiv((long long)ZB_ * LD_ * HD_, 256), 256, 0, stream>>>(vmean, O2);
  k_toprow<<<dim3(NT_, ZB_), 256, 0, stream>>>(Qbf, Kbf, Vtbf, topidx, O2);

  // ---- per-branch output projection Z_i = merge(O_i) @ Wo (WMMA) ----
  float* Os[5] = { O0, O1, O2, O3, O4 };
  for (int i = 0; i < 5; ++i) {
    k_merge<<<cdiv((long long)B_ * LD_ * D_, 256), 256, 0, stream>>>(Os[i], Ybf);
    gemm(Ybf, WoT, Zp[i], B_ * LD_, D_, D_, D_, D_, D_, 1, 1, 1, 0, 0, 0, 0, 0, 0, 0);
  }

  // ---- residual + RMSNorm + softmax(alphas) mixture ----
  k_finalize<<<B_ * LD_, 256, 0, stream>>>(dec, Zp[0], Zp[1], Zp[2], Zp[3], Zp[4],
                                           rmsw, w6, out);
}
